// HierarchicalMoE_1520418423053
// MI455X (gfx1250) — compile-verified, matrix operand order checked by reading, not run on hardware
//
#include <hip/hip_runtime.h>
#include <hip/hip_bf16.h>
#include <math.h>
#include <stdint.h>

// Problem dims (fixed by the reference)
#define T_TOK 8192   // B*S = 4*2048
#define H_DIM 1024
#define I_DIM 1024
#define NEXP  8

typedef __bf16 bf16;
typedef __attribute__((ext_vector_type(16))) __bf16 v16bf;
typedef __attribute__((ext_vector_type(8)))  __bf16 v8bf;
typedef __attribute__((ext_vector_type(8)))  float  v8f;

// Pointer types matching the async builtin's prototype:
//   (v4i addrspace(1)*, v4i addrspace(3)*, Ii offset, Ii cpol)
typedef int vsi4 __attribute__((vector_size(16)));
typedef __attribute__((address_space(1))) vsi4 as1_vsi4;
typedef __attribute__((address_space(3))) vsi4 as3_vsi4;

// ---------------------------------------------------------------------------
// Async global->LDS copy (16B per lane). gfx1250 GLOBAL_LOAD_ASYNC_TO_LDS_B128,
// tracked by ASYNCcnt. Guarded: falls back to a synchronous copy if the
// builtin is absent on this toolchain.
// ---------------------------------------------------------------------------
#if defined(__has_builtin) && __has_builtin(__builtin_amdgcn_global_load_async_to_lds_b128)
#define HAVE_ASYNC 1
static __device__ __forceinline__ void async_cp16(const bf16* g, bf16* l) {
  // generic->AS1 and generic->AS3 via integer casts (LDS offset = addr[31:0])
  __builtin_amdgcn_global_load_async_to_lds_b128(
      (as1_vsi4*)(uintptr_t)g,
      (as3_vsi4*)(uint32_t)(uintptr_t)l,
      0, 0);
}
#if defined(__has_builtin) && __has_builtin(__builtin_amdgcn_s_wait_asynccnt)
#define WAIT_ASYNC(n) __builtin_amdgcn_s_wait_asynccnt(n)
#else
#define WAIT_ASYNC(n) asm volatile("s_wait_asynccnt %0" ::"n"(n) : "memory")
#endif
#else
#define HAVE_ASYNC 0
static __device__ __forceinline__ void async_cp16(const bf16* g, bf16* l) {
  *(v8bf*)l = *(const v8bf*)g;   // synchronous fallback
}
#define WAIT_ASYNC(n) ((void)0)
#endif

// Load one 16x32 bf16 WMMA fragment from LDS (K-major rows, caller applied
// the lane's half-offset). Two 16B ds loads -> ds_load_b128 pairs.
static __device__ __forceinline__ v16bf lds_frag(const bf16* p) {
  union { v16bf v; v8bf h[2]; } u;
  u.h[0] = *(const v8bf*)(p);
  u.h[1] = *(const v8bf*)(p + 16);
  return u.v;
}

// ---------------------------------------------------------------------------
// Router: logits = x @ w_router, softmax, top-2, renormalize.
// One wave (32 lanes) per token. Output: dense [T][8] combine weights.
// ---------------------------------------------------------------------------
__global__ __launch_bounds__(256) void moe_router(const float* __restrict__ x,
                                                  const float* __restrict__ wr,
                                                  float* __restrict__ wroute) {
  const int gid  = blockIdx.x * blockDim.x + threadIdx.x;
  const int t    = gid >> 5;
  const int lane = gid & 31;
  if (t >= T_TOK) return;

  float acc[NEXP];
#pragma unroll
  for (int e = 0; e < NEXP; ++e) acc[e] = 0.0f;

  const float* xr = x + (size_t)t * H_DIM;
  for (int h = lane; h < H_DIM; h += 32) {
    const float xv = xr[h];
#pragma unroll
    for (int e = 0; e < NEXP; ++e) acc[e] += xv * wr[h * NEXP + e];
  }
#pragma unroll
  for (int e = 0; e < NEXP; ++e) {
#pragma unroll
    for (int off = 16; off > 0; off >>= 1) acc[e] += __shfl_down(acc[e], off, 32);
  }

  if (lane == 0) {
    float m = acc[0];
#pragma unroll
    for (int e = 1; e < NEXP; ++e) m = fmaxf(m, acc[e]);
    float p[NEXP];
#pragma unroll
    for (int e = 0; e < NEXP; ++e) p[e] = __expf(acc[e] - m);
    int i1 = 0;
#pragma unroll
    for (int e = 1; e < NEXP; ++e) if (p[e] > p[i1]) i1 = e;
    int i2 = (i1 == 0) ? 1 : 0;
#pragma unroll
    for (int e = 0; e < NEXP; ++e) if (e != i1 && p[e] > p[i2]) i2 = e;
    const float s = p[i1] + p[i2];
#pragma unroll
    for (int e = 0; e < NEXP; ++e) {
      float w = 0.0f;
      if (e == i1) w = p[i1] / s;
      else if (e == i2) w = p[i2] / s;
      wroute[(size_t)t * NEXP + e] = w;
    }
  }
}

// ---------------------------------------------------------------------------
// Elementwise fp32 -> bf16
// ---------------------------------------------------------------------------
__global__ __launch_bounds__(256) void cvt_f32_bf16(const float* __restrict__ src,
                                                    bf16* __restrict__ dst, int n) {
  const int i = blockIdx.x * blockDim.x + threadIdx.x;
  if (i < n) dst[i] = (bf16)src[i];
}

// ---------------------------------------------------------------------------
// Per-expert 1024x1024 transpose + fp32->bf16 (LDS-tiled, coalesced)
// ---------------------------------------------------------------------------
__global__ __launch_bounds__(256) void tcvt_1024(const float* __restrict__ src,
                                                 bf16* __restrict__ dst) {
  __shared__ bf16 tile[32][33];
  const int e  = blockIdx.z;
  const size_t mo = (size_t)e * H_DIM * I_DIM;
  const int c0 = blockIdx.x * 32;
  const int r0 = blockIdx.y * 32;
#pragma unroll
  for (int i = threadIdx.y; i < 32; i += 8)
    tile[i][threadIdx.x] = (bf16)src[mo + (size_t)(r0 + i) * 1024 + c0 + threadIdx.x];
  __syncthreads();
#pragma unroll
  for (int i = threadIdx.y; i < 32; i += 8)
    dst[mo + (size_t)(c0 + i) * 1024 + r0 + threadIdx.x] = tile[threadIdx.x][i];
}

// ---------------------------------------------------------------------------
// LDS staging geometry (shared by both GEMM kernels)
//   Workgroup tile: 128 M x 64 N, K staged 64 at a time, double buffered.
//   Row pitch padded 64 -> 72 elems (144B) => conflict-free ds_load_b128.
// ---------------------------------------------------------------------------
#define KB      64
#define LDP     72
#define A_ELEMS (128 * LDP)            // 9216 elems (18KB)
#define B_ELEMS (64 * LDP)             // 4608 elems (9KB)
#define STAGE_GU (A_ELEMS + 2 * B_ELEMS)   // gate+up: 18432 elems (36KB)
#define STAGE_D  (A_ELEMS + B_ELEMS)       // down:    13824 elems (27KB)

#define WMMA_BF16(a, b, c) \
  __builtin_amdgcn_wmma_f32_16x16x32_bf16(false, (a), false, (b), (short)0, (c), false, false)

// ---------------------------------------------------------------------------
// Fused gate+up GEMM for one expert:
//   hb = silu(x@Wg) * (x@Wu);  A = xb [T][H], B = Wgt/Wut [I][H] (K-major rows)
// Grid: (I/64, T/128), block 256 = 8 waves. Double-buffered async LDS stages.
// ---------------------------------------------------------------------------
__global__ __launch_bounds__(256) void moe_gateup(const bf16* __restrict__ xb,
                                                  const bf16* __restrict__ wgt,
                                                  const bf16* __restrict__ wut,
                                                  bf16* __restrict__ hb) {
  __shared__ bf16 smem[2 * STAGE_GU];
  const int t    = threadIdx.x;
  const int lane = t & 31;
  const int wave = t >> 5;
  const int r = lane & 15, half = lane >> 4;
  const int m_base = blockIdx.y * 128;
  const int n_base = blockIdx.x * 64;

  v8f accg[4], accu[4];
  const v8f vzero = {};
#pragma unroll
  for (int j = 0; j < 4; ++j) { accg[j] = vzero; accu[j] = vzero; }

  auto load_stage = [&](int s, int b) {
    const int k0 = s * KB;
    bf16* dst = smem + b * STAGE_GU;
#pragma unroll
    for (int i = 0; i < 4; ++i) {               // A: 1024 x 16B chunks
      const int c = t + 256 * i;
      const int row = c >> 3, kc = c & 7;
      async_cp16(xb + (size_t)(m_base + row) * H_DIM + k0 + kc * 8,
                 dst + row * LDP + kc * 8);
    }
#pragma unroll
    for (int i = 0; i < 2; ++i) {               // Bg + Bu: 512 chunks each
      const int c = t + 256 * i;
      const int row = c >> 3, kc = c & 7;
      async_cp16(wgt + (size_t)(n_base + row) * H_DIM + k0 + kc * 8,
                 dst + A_ELEMS + row * LDP + kc * 8);
      async_cp16(wut + (size_t)(n_base + row) * H_DIM + k0 + kc * 8,
                 dst + A_ELEMS + B_ELEMS + row * LDP + kc * 8);
    }
  };

  load_stage(0, 0);
  int buf = 0;
  const int nstage = H_DIM / KB;   // 16
  for (int s = 0; s < nstage; ++s) {
    if (s + 1 < nstage) {
      load_stage(s + 1, buf ^ 1);  // prefetch next stage (8 asyncs/thread)
      WAIT_ASYNC(8);               // stage s landed; next stage in flight
    } else {
      WAIT_ASYNC(0);
    }
    __syncthreads();
    const bf16* As  = smem + buf * STAGE_GU + (wave * 16 + r) * LDP + half * 8;
    const bf16* Bgs = smem + buf * STAGE_GU + A_ELEMS + r * LDP + half * 8;
    const bf16* Bus = Bgs + B_ELEMS;
#pragma unroll
    for (int kk = 0; kk < KB; kk += 32) {
      const v16bf a = lds_frag(As + kk);
#pragma unroll
      for (int j = 0; j < 4; ++j) {
        const v16bf bg = lds_frag(Bgs + j * 16 * LDP + kk);
        const v16bf bu = lds_frag(Bus + j * 16 * LDP + kk);
        accg[j] = WMMA_BF16(a, bg, accg[j]);
        accu[j] = WMMA_BF16(a, bu, accu[j]);
      }
    }
    __syncthreads();
    buf ^= 1;
  }

  // C/D layout: VGPR i -> row half*8+i, lane&15 -> col
#pragma unroll
  for (int j = 0; j < 4; ++j) {
#pragma unroll
    for (int i = 0; i < 8; ++i) {
      const float g = accg[j][i];
      const float u = accu[j][i];
      const float sv = (g / (1.0f + __expf(-g))) * u;   // silu(g) * u
      const int row = m_base + wave * 16 + half * 8 + i;
      const int col = n_base + j * 16 + r;
      hb[(size_t)row * I_DIM + col] = (bf16)sv;
    }
  }
}

// ---------------------------------------------------------------------------
// Down GEMM for one expert: out[t][h] (+)= (hb @ Wd)[t][h] * wroute[t][e]
// A = hb [T][I], B = Wdt [H][I] (K-major rows). Same staging scheme.
// ---------------------------------------------------------------------------
__global__ __launch_bounds__(256) void moe_down(const bf16* __restrict__ hbuf,
                                                const bf16* __restrict__ wdt,
                                                const float* __restrict__ wroute,
                                                float* __restrict__ out,
                                                int expert, int accum) {
  __shared__ bf16 smem[2 * STAGE_D];
  const int t    = threadIdx.x;
  const int lane = t & 31;
  const int wave = t >> 5;
  const int r = lane & 15, half = lane >> 4;
  const int m_base = blockIdx.y * 128;
  const int n_base = blockIdx.x * 64;

  v8f acc[4];
  const v8f vzero = {};
#pragma unroll
  for (int j = 0; j < 4; ++j) acc[j] = vzero;

  auto load_stage = [&](int s, int b) {
    const int k0 = s * KB;
    bf16* dst = smem + b * STAGE_D;
#pragma unroll
    for (int i = 0; i < 4; ++i) {
      const int c = t + 256 * i;
      const int row = c >> 3, kc = c & 7;
      async_cp16(hbuf + (size_t)(m_base + row) * I_DIM + k0 + kc * 8,
                 dst + row * LDP + kc * 8);
    }
#pragma unroll
    for (int i = 0; i < 2; ++i) {
      const int c = t + 256 * i;
      const int row = c >> 3, kc = c & 7;
      async_cp16(wdt + (size_t)(n_base + row) * I_DIM + k0 + kc * 8,
                 dst + A_ELEMS + row * LDP + kc * 8);
    }
  };

  load_stage(0, 0);
  int buf = 0;
  const int nstage = I_DIM / KB;   // 16
  for (int s = 0; s < nstage; ++s) {
    if (s + 1 < nstage) {
      load_stage(s + 1, buf ^ 1);  // 6 asyncs/thread
      WAIT_ASYNC(6);
    } else {
      WAIT_ASYNC(0);
    }
    __syncthreads();
    const bf16* As = smem + buf * STAGE_D + (wave * 16 + r) * LDP + half * 8;
    const bf16* Bs = smem + buf * STAGE_D + A_ELEMS + r * LDP + half * 8;
#pragma unroll
    for (int kk = 0; kk < KB; kk += 32) {
      const v16bf a = lds_frag(As + kk);
#pragma unroll
      for (int j = 0; j < 4; ++j) {
        const v16bf b = lds_frag(Bs + j * 16 * LDP + kk);
        acc[j] = WMMA_BF16(a, b, acc[j]);
      }
    }
    __syncthreads();
    buf ^= 1;
  }

#pragma unroll
  for (int j = 0; j < 4; ++j) {
#pragma unroll
    for (int i = 0; i < 8; ++i) {
      const int row = m_base + wave * 16 + half * 8 + i;
      const int col = n_base + j * 16 + r;
      const float wtok = wroute[(size_t)row * NEXP + expert];
      const float v = acc[j][i] * wtok;
      const size_t o = (size_t)row * H_DIM + col;
      out[o] = accum ? (out[o] + v) : v;
    }
  }
}

// ---------------------------------------------------------------------------
// Host-side orchestration (all on `stream`; workspace carved from d_ws)
// ---------------------------------------------------------------------------
extern "C" void kernel_launch(void* const* d_in, const int* in_sizes, int n_in,
                              void* d_out, int out_size, void* d_ws, size_t ws_size,
                              hipStream_t stream) {
  (void)in_sizes; (void)n_in; (void)out_size; (void)ws_size;
  const float* x  = (const float*)d_in[0];   // [B,S,H]
  const float* wr = (const float*)d_in[1];   // [H,E]
  const float* Wg = (const float*)d_in[2];   // [E,H,I]
  const float* Wu = (const float*)d_in[3];   // [E,H,I]
  const float* Wd = (const float*)d_in[4];   // [E,I,H]
  float* out = (float*)d_out;                // [B,S,H] fp32

  char* ws = (char*)d_ws;
  size_t off = 0;
  auto carve = [&](size_t bytes) -> void* {
    void* p = (void*)(ws + off);
    off += (bytes + 255) & ~(size_t)255;
    return p;
  };
  const size_t EHI = (size_t)NEXP * H_DIM * I_DIM;
  float* wroute = (float*)carve((size_t)T_TOK * NEXP * sizeof(float));
  bf16*  xb     = (bf16*) carve((size_t)T_TOK * H_DIM * sizeof(bf16));
  bf16*  hbuf   = (bf16*) carve((size_t)T_TOK * I_DIM * sizeof(bf16));
  bf16*  wgt    = (bf16*) carve(EHI * sizeof(bf16));
  bf16*  wut    = (bf16*) carve(EHI * sizeof(bf16));
  bf16*  wdt    = (bf16*) carve(EHI * sizeof(bf16));

  // 1) Router
  moe_router<<<T_TOK / 8, 256, 0, stream>>>(x, wr, wroute);

  // 2) Precision conversion / weight transposition
  {
    const int n = T_TOK * H_DIM;
    cvt_f32_bf16<<<(n + 255) / 256, 256, 0, stream>>>(x, xb, n);
  }
  {
    dim3 tg(32, 32, NEXP), tb(32, 8);
    tcvt_1024<<<tg, tb, 0, stream>>>(Wg, wgt);  // [E,H,I] -> [E,I,H]
    tcvt_1024<<<tg, tb, 0, stream>>>(Wu, wut);  // [E,H,I] -> [E,I,H]
    tcvt_1024<<<tg, tb, 0, stream>>>(Wd, wdt);  // [E,I,H] -> [E,H,I]
  }

  // 3) Dense per-expert FFN; expert 0 writes `out`, the rest accumulate.
  const dim3 ggrid(I_DIM / 64, T_TOK / 128);   // (16, 64)
  const dim3 dgrid(H_DIM / 64, T_TOK / 128);   // (16, 64)
  for (int e = 0; e < NEXP; ++e) {
    const size_t wo = (size_t)e * H_DIM * I_DIM;
    moe_gateup<<<ggrid, 256, 0, stream>>>(xb, wgt + wo, wut + wo, hbuf);
    moe_down  <<<dgrid, 256, 0, stream>>>(hbuf, wdt + wo, wroute, out, e, e > 0 ? 1 : 0);
  }
}
